// NeighborSelection_76330158784614
// MI455X (gfx1250) — matrix operand order repaired and checked by reading, not compiled, not run on hardware
//
#include <hip/hip_runtime.h>
#include <hip/hip_bf16.h>
#include <math.h>

typedef __attribute__((ext_vector_type(16))) _Float16 v16h;
typedef __attribute__((ext_vector_type(8)))  float    v8f;

#define DFEAT 256
#define MNB   64
#define KSEL  3
#define KCHUNKS (DFEAT / 32)   // 8 WMMA k-steps

// ---------------------------------------------------------------------------
// Kernel 1: fused projection  s_self = X·wa + b,  s_all = X·wb
// One wave32 per 16-node tile. A = 16x32 f16 slice of X, B = 32x16 with
// wa in column 0, wb in column 1 (rest zero). B tiles are loop-invariant and
// hoisted (64 VGPRs). The A-stream is software-pipelined one chunk deep with
// distinct float4 staging registers so the 4 b128 loads of chunk kc+1 issue
// as a clause and fly while chunk kc is converted and fed to v_wmma.
// Memory-bound: streams X once (205 MB at 23.3 TB/s ~ 9 us floor).
// ---------------------------------------------------------------------------
__global__ __launch_bounds__(256)
void gat_score_wmma(const float* __restrict__ X, const float* __restrict__ W,
                    const float* __restrict__ bias,
                    float* __restrict__ s_self, float* __restrict__ s_all, int N)
{
    const int lane = threadIdx.x & 31;
    const int wv   = threadIdx.x >> 5;
    const int base = (blockIdx.x * (blockDim.x >> 5) + wv) << 4;   // 16 nodes/wave
    if (base >= N) return;                 // wave-uniform guard (N % 16 == 0)

    const int m  = lane & 15;              // row within tile (A) / column n (B,C)
    const int hi = lane >> 4;              // half-select for K layout

    // ---- Hoist all B tiles (loop-invariant weights) -----------------------
    // B 32x16 f16 layout: lane = n + 16*hi; per lane K = {8hi..8hi+7, 16+8hi..16+8hi+7}.
    // Column 0 = wa, column 1 = wb, columns 2..15 = 0.
    v16h btile[KCHUNKS];
    #pragma unroll
    for (int kc = 0; kc < KCHUNKS; ++kc) {
        v16h bc = {};
        if (m < 2) {
            const float* wp = W + m * DFEAT + kc * 32 + hi * 8;
            #pragma unroll
            for (int e = 0; e < 16; ++e) bc[e] = (_Float16)wp[e < 8 ? e : e + 8];
        }
        btile[kc] = bc;
    }

    const float* xrow = X + (size_t)(base + m) * DFEAT;

    // A 16x32 f16 layout: lanes 0-15 row m hold K = {0..7, 16..23},
    // lanes 16-31 row m hold K = {8..15, 24..31}  -> per lane, per chunk:
    // 2x float4 at (kc*32 + hi*8) and 2x float4 at (+16).
    float4 c0, c1, c2, c3;     // current chunk staging (f32)
    float4 n0, n1, n2, n3;     // next chunk staging (f32)

    {
        const float4* xp = (const float4*)(xrow + 0 * 32 + hi * 8);
        c0 = xp[0]; c1 = xp[1]; c2 = xp[4]; c3 = xp[5];
    }

    v8f acc = {};
    #pragma unroll
    for (int kc = 0; kc < KCHUNKS; ++kc) {
        if (kc + 1 < KCHUNKS) {            // prefetch next chunk (clause of 4)
            const float4* xp = (const float4*)(xrow + (kc + 1) * 32 + hi * 8);
            n0 = xp[0]; n1 = xp[1]; n2 = xp[4]; n3 = xp[5];
        }

        v16h a;
        a[0]  = (_Float16)c0.x; a[1]  = (_Float16)c0.y;
        a[2]  = (_Float16)c0.z; a[3]  = (_Float16)c0.w;
        a[4]  = (_Float16)c1.x; a[5]  = (_Float16)c1.y;
        a[6]  = (_Float16)c1.z; a[7]  = (_Float16)c1.w;
        a[8]  = (_Float16)c2.x; a[9]  = (_Float16)c2.y;
        a[10] = (_Float16)c2.z; a[11] = (_Float16)c2.w;
        a[12] = (_Float16)c3.x; a[13] = (_Float16)c3.y;
        a[14] = (_Float16)c3.z; a[15] = (_Float16)c3.w;

        acc = __builtin_amdgcn_wmma_f32_16x16x32_f16(
                  /*neg_a=*/false, a, /*neg_b=*/false, btile[kc],
                  /*c_mod=*/(short)0, acc, /*reuse_a=*/false, /*reuse_b=*/false);

        c0 = n0; c1 = n1; c2 = n2; c3 = n3;   // rotate pipeline (reg renames)
    }

    // C 16x16 f32 layout: lane = n + 16*(m>=8); VGPR i holds row (8*hi + i).
    // Only columns 0 (s_self) and 1 (s_all) carry data.
    if (m < 2) {
        const float add = (m == 0) ? bias[0] : 0.0f;
        float* dst = ((m == 0) ? s_self : s_all) + base + hi * 8;
        #pragma unroll
        for (int i = 0; i < 8; ++i) dst[i] = acc[i] + add;
    }
}

// ---------------------------------------------------------------------------
// Kernel 2: one wave32 per node. Each lane scores 2 of the 64 neighbors
// (single b128 load of two int64 ids; the s_all gather table is only 800 KB
// so it is L2-resident), then 3 rounds of wave-wide argmax via shfl_xor
// (stable: lowest index wins ties, matching jax.lax.top_k).
// ---------------------------------------------------------------------------
__global__ __launch_bounds__(256)
void gat_topk(const long long* __restrict__ nbr, const long long* __restrict__ cnt,
              const float* __restrict__ s_self, const float* __restrict__ s_all,
              float* __restrict__ out_vals, float* __restrict__ out_sel, int N)
{
    const int lane = threadIdx.x & 31;
    const int wv   = threadIdx.x >> 5;
    const int node = blockIdx.x * (blockDim.x >> 5) + wv;
    if (node >= N) return;

    const float     ss = s_self[node];
    const long long c  = cnt[node];

    // Two consecutive neighbor ids per lane -> one global_load_b128.
    const longlong2* np =
        (const longlong2*)(nbr + (size_t)node * MNB) + lane;
    const longlong2 nn = *np;

    float val[2];
    float nbf[2];
    const long long njs[2] = { nn.x, nn.y };
    #pragma unroll
    for (int t = 0; t < 2; ++t) {
        const int j        = lane * 2 + t;
        const long long nj = njs[t];
        const float x  = ss + s_all[nj];
        const float lr = (x > 0.0f) ? x : 0.01f * x;   // leaky_relu, slope 0.01
        float sc = __expf(lr);
        if ((long long)j >= c) sc = -INFINITY;
        val[t] = sc;
        nbf[t] = (float)nj;                            // < 2^24, exact in f32
    }

    #pragma unroll
    for (int r = 0; r < KSEL; ++r) {
        const int tb = (val[1] > val[0]) ? 1 : 0;      // local best (tie -> t=0)
        float bv = val[tb];
        int   bj = lane * 2 + tb;
        #pragma unroll
        for (int off = 16; off > 0; off >>= 1) {
            const float ov = __shfl_xor(bv, off, 32);
            const int   oj = __shfl_xor(bj, off, 32);
            if (ov > bv || (ov == bv && oj < bj)) { bv = ov; bj = oj; }
        }
        // bj/bv now wave-uniform; fetch the winner's neighbor id.
        const float sel = __shfl(nbf[bj & 1], bj >> 1, 32);
        if (lane == 0) {
            out_vals[node * KSEL + r] = bv;
            out_sel [node * KSEL + r] = sel;
        }
        if (lane == (bj >> 1)) val[bj & 1] = -INFINITY; // knock out winner
    }
}

// ---------------------------------------------------------------------------
extern "C" void kernel_launch(void* const* d_in, const int* in_sizes, int n_in,
                              void* d_out, int out_size, void* d_ws, size_t ws_size,
                              hipStream_t stream)
{
    const float*     X   = (const float*)d_in[0];       // (N, 256) f32
    const long long* nbr = (const long long*)d_in[1];   // (N, 64) i64
    const long long* cnt = (const long long*)d_in[2];   // (N,) i64
    const float*     W   = (const float*)d_in[3];       // (512,) f32
    const float*     b   = (const float*)d_in[4];       // (1,) f32

    const int N = in_sizes[2];

    float* s_self = (float*)d_ws;
    float* s_all  = s_self + N;

    float* out_vals = (float*)d_out;
    float* out_sel  = out_vals + (size_t)N * KSEL;

    {
        const int nodesPerBlock = (256 / 32) * 16;      // 8 waves x 16 nodes
        const int grid = (N + nodesPerBlock - 1) / nodesPerBlock;
        hipLaunchKernelGGL(gat_score_wmma, dim3(grid), dim3(256), 0, stream,
                           X, W, b, s_self, s_all, N);
    }
    {
        const int wavesPerBlock = 256 / 32;             // 1 node per wave
        const int grid = (N + wavesPerBlock - 1) / wavesPerBlock;
        hipLaunchKernelGGL(gat_topk, dim3(grid), dim3(256), 0, stream,
                           nbr, cnt, s_self, s_all, out_vals, out_sel, N);
    }
}
